// SelfMHA_61503931678846
// MI455X (gfx1250) — matrix operand order repaired and checked by reading, not compiled, not run on hardware
//
#include <hip/hip_runtime.h>

typedef __attribute__((ext_vector_type(16))) __bf16 v16bf;
typedef __attribute__((ext_vector_type(8)))  float  v8f;

#define NHEADS 12
#define SEQ    2048
#define DMODEL 768
#define DHEAD  64
#define BATCH  2

// Async global->LDS path (gfx1250). Guarded so a missing builtin falls back
// to plain dword copies instead of breaking the build.
#if __has_builtin(__builtin_amdgcn_global_load_async_to_lds_b128) && \
    __has_builtin(__builtin_amdgcn_s_wait_asynccnt)
#define HAVE_ASYNC_LDS 1
typedef int i32x4 __attribute__((vector_size(16)));
typedef __attribute__((address_space(1))) i32x4* async_gptr;
typedef __attribute__((address_space(3))) i32x4* async_lptr;
#else
#define HAVE_ASYNC_LDS 0
#endif

// ---------------------------------------------------------------------------
// WMMA helpers (layouts per CDNA5 ISA 7.12.2)
// ---------------------------------------------------------------------------
__device__ __forceinline__ v8f wmma_bf16(v16bf a, v16bf b, v8f c) {
  return __builtin_amdgcn_wmma_f32_16x16x32_bf16(
      /*neg_a=*/false, a, /*neg_b=*/false, b,
      /*c_mod=*/(short)0, c, /*reuse_a=*/false, /*reuse_b=*/false);
}

// A-matrix 16x32 bf16 from LDS, rows row0..row0+15, K-slice k0..k0+31, ld elems.
// Lane L: m = L&15, half = L>>4; element j -> k = j + 16*half.
__device__ __forceinline__ v16bf load_a_frag(const __bf16* lds, int row0, int k0, int ld) {
  int l = threadIdx.x & 31;
  int hf = l >> 4;
  const __bf16* rp = lds + (size_t)(row0 + (l & 15)) * ld + k0 + hf * 16;
  v16bf a;
#pragma unroll
  for (int j = 0; j < 16; ++j) a[j] = rp[j];
  return a;
}

// B-matrix 32x16 from LDS stored row-major [k][ld]. Lane L: n = L&15;
// lanes 0-15 cover K=k0..k0+15, lanes 16-31 cover K=k0+16..k0+31.
__device__ __forceinline__ v16bf load_b_frag_km(const __bf16* lds, int k0, int n0, int ld) {
  int l = threadIdx.x & 31;
  const __bf16* p = lds + (size_t)(k0 + (l >> 4) * 16) * ld + n0 + (l & 15);
  v16bf b;
#pragma unroll
  for (int j = 0; j < 16; ++j) b[j] = p[(size_t)j * ld];
  return b;
}

// B-matrix 32x16 where LDS storage is [n][ld] with inner dim = k (B = stored^T).
__device__ __forceinline__ v16bf load_b_frag_nk(const __bf16* lds, int n0, int k0, int ld) {
  int l = threadIdx.x & 31;
  const __bf16* p = lds + (size_t)(n0 + (l & 15)) * ld + k0 + (l >> 4) * 16;
  v16bf b;
#pragma unroll
  for (int j = 0; j < 16; ++j) b[j] = p[j];
  return b;
}

// ---------------------------------------------------------------------------
// Kernel 1: qkv = xs @ Wqkv  (fp32 in, bf16 WMMA, scatter to per-head Q/K/V)
// M=4096 N=2304 K=768; block tile 128x128, 4 waves each doing 64x64.
// ---------------------------------------------------------------------------
__global__ __launch_bounds__(128) void qkv_gemm_kernel(
    const float* __restrict__ xs, const float* __restrict__ Wqkv,
    __bf16* __restrict__ q_ws, __bf16* __restrict__ k_ws, __bf16* __restrict__ v_ws) {
  const int LDA = 40, LDB = 132;
  __shared__ __bf16 As[128 * 40];
  __shared__ __bf16 Bs[32 * 132];

  int tid = threadIdx.x, wave = tid >> 5, lane = tid & 31;
  int wm = wave >> 1, wn = wave & 1;          // 2x2 wave grid, 64x64 per wave
  int m_blk = blockIdx.y * 128, n_blk = blockIdx.x * 128;

  v8f zero = {0.f, 0.f, 0.f, 0.f, 0.f, 0.f, 0.f, 0.f};
  v8f acc[4][4];
#pragma unroll
  for (int i = 0; i < 4; ++i)
#pragma unroll
    for (int j = 0; j < 4; ++j) acc[i][j] = zero;

  for (int kt = 0; kt < DMODEL / 32; ++kt) {
    __syncthreads();
    if (kt + 1 < DMODEL / 32) {  // hide HBM latency on the next K-slice
      __builtin_prefetch(xs + (size_t)(m_blk + tid) * DMODEL + (kt + 1) * 32, 0, 1);
      __builtin_prefetch(Wqkv + (size_t)((kt + 1) * 32 + (tid >> 2)) * (3 * DMODEL) +
                             n_blk + ((tid & 3) << 5), 0, 1);
    }
#pragma unroll
    for (int it = 0; it < 32; ++it) {        // A tile 128x32
      int idx = tid + it * 128;
      int r = idx >> 5, c = idx & 31;
      As[r * LDA + c] = (__bf16)xs[(size_t)(m_blk + r) * DMODEL + kt * 32 + c];
    }
#pragma unroll
    for (int it = 0; it < 32; ++it) {        // B tile 32x128
      int idx = tid + it * 128;
      int r = idx >> 7, c = idx & 127;
      Bs[r * LDB + c] = (__bf16)Wqkv[(size_t)(kt * 32 + r) * (3 * DMODEL) + n_blk + c];
    }
    __syncthreads();

    v16bf af[4];
#pragma unroll
    for (int mf = 0; mf < 4; ++mf) af[mf] = load_a_frag(As, wm * 64 + mf * 16, 0, LDA);
#pragma unroll
    for (int nf = 0; nf < 4; ++nf) {
      v16bf bfr = load_b_frag_km(Bs, 0, wn * 64 + nf * 16, LDB);
#pragma unroll
      for (int mf = 0; mf < 4; ++mf) acc[mf][nf] = wmma_bf16(af[mf], bfr, acc[mf][nf]);
    }
  }

  int hf = lane >> 4, nl = lane & 15;
#pragma unroll
  for (int mf = 0; mf < 4; ++mf)
#pragma unroll
    for (int nf = 0; nf < 4; ++nf)
#pragma unroll
      for (int r = 0; r < 8; ++r) {
        int m = m_blk + wm * 64 + mf * 16 + hf * 8 + r;   // b*S + s
        int n = n_blk + wn * 64 + nf * 16 + nl;           // column of [Q|K|V]
        int which = n / DMODEL;
        int d768 = n - which * DMODEL;                    // d768 = dd*H + h
        int h = d768 % NHEADS, dd = d768 / NHEADS;
        int b = m >> 11, s = m & (SEQ - 1);
        size_t dst = (((size_t)(b * NHEADS + h)) * SEQ + s) * DHEAD + dd;
        __bf16 val = (__bf16)acc[mf][nf][r];
        (which == 0 ? q_ws : (which == 1 ? k_ws : v_ws))[dst] = val;
      }
}

// ---------------------------------------------------------------------------
// Kernel 2: flash attention per (b, h, 64-row query tile); 4 waves x 16 rows.
// ---------------------------------------------------------------------------
__global__ __launch_bounds__(128) void attn_kernel(
    const __bf16* __restrict__ q_ws, const __bf16* __restrict__ k_ws,
    const __bf16* __restrict__ v_ws, const unsigned char* __restrict__ mask,
    __bf16* __restrict__ atn_ws) {
  __shared__ __bf16 Qs[64 * DHEAD];
  __shared__ __bf16 Ks[64 * DHEAD];
  __shared__ __bf16 Vs[64 * DHEAD];
  __shared__ __bf16 Ps[4][16 * DHEAD];

  int tid = threadIdx.x, wave = tid >> 5, lane = tid & 31;
  int hf = lane >> 4, nl = lane & 15;
  int bh = blockIdx.x >> 5;                  // b*NHEADS + h
  int qt = blockIdx.x & 31;                  // query tile (64 rows)
  int b = bh / NHEADS, h = bh - b * NHEADS;

  const __bf16* Qg = q_ws + (size_t)bh * SEQ * DHEAD;
  const __bf16* Kg = k_ws + (size_t)bh * SEQ * DHEAD;
  const __bf16* Vg = v_ws + (size_t)bh * SEQ * DHEAD;

#if HAVE_ASYNC_LDS
  {  // Q tile: 8192 B via async b128 copies (ASYNCcnt-tracked, no VGPR data)
    async_gptr g = (async_gptr)(Qg + (size_t)qt * 64 * DHEAD);
    async_lptr l = (async_lptr)Qs;
#pragma unroll
    for (int it = 0; it < 4; ++it) {
      int e = tid + it * 128;   // 16B elements
      __builtin_amdgcn_global_load_async_to_lds_b128(g + e, l + e, 0, 0);
    }
  }
#else
  {
    const unsigned int* src = (const unsigned int*)(Qg + (size_t)qt * 64 * DHEAD);
    unsigned int* dst = (unsigned int*)Qs;
#pragma unroll
    for (int it = 0; it < 16; ++it) dst[tid + it * 128] = src[tid + it * 128];
  }
#endif

  float m_i[8], l_i[8];
  v8f zero = {0.f, 0.f, 0.f, 0.f, 0.f, 0.f, 0.f, 0.f};
  v8f o_acc[4];
#pragma unroll
  for (int r = 0; r < 8; ++r) { m_i[r] = -1e30f; l_i[r] = 0.f; }
#pragma unroll
  for (int nf = 0; nf < 4; ++nf) o_acc[nf] = zero;

  for (int kt = 0; kt < SEQ / 64; ++kt) {
    __syncthreads();
#if HAVE_ASYNC_LDS
    {
      async_gptr gk = (async_gptr)(Kg + (size_t)kt * 64 * DHEAD);
      async_gptr gv = (async_gptr)(Vg + (size_t)kt * 64 * DHEAD);
      async_lptr lk = (async_lptr)Ks;
      async_lptr lv = (async_lptr)Vs;
#pragma unroll
      for (int it = 0; it < 4; ++it) {
        int e = tid + it * 128;
        __builtin_amdgcn_global_load_async_to_lds_b128(gk + e, lk + e, 0, 0);
        __builtin_amdgcn_global_load_async_to_lds_b128(gv + e, lv + e, 0, 0);
      }
    }
    __builtin_amdgcn_s_wait_asynccnt(0);
#else
    {
      const unsigned int* ks = (const unsigned int*)(Kg + (size_t)kt * 64 * DHEAD);
      const unsigned int* vs = (const unsigned int*)(Vg + (size_t)kt * 64 * DHEAD);
      unsigned int* kd = (unsigned int*)Ks;
      unsigned int* vd = (unsigned int*)Vs;
#pragma unroll
      for (int it = 0; it < 16; ++it) {
        kd[tid + it * 128] = ks[tid + it * 128];
        vd[tid + it * 128] = vs[tid + it * 128];
      }
    }
#endif
    __syncthreads();

    // S = Q(16x64) . K^T(64x64) for this wave's 16 rows
    v8f sc[4];
    v16bf qa0 = load_a_frag(Qs, wave * 16, 0, DHEAD);
    v16bf qa1 = load_a_frag(Qs, wave * 16, 32, DHEAD);
#pragma unroll
    for (int nf = 0; nf < 4; ++nf) {
      v16bf kb0 = load_b_frag_nk(Ks, nf * 16, 0, DHEAD);
      v16bf kb1 = load_b_frag_nk(Ks, nf * 16, 32, DHEAD);
      sc[nf] = wmma_bf16(qa0, kb0, zero);
      sc[nf] = wmma_bf16(qa1, kb1, sc[nf]);
    }

    // scale by 1/sqrt(dh)=0.125, then mask -> -1e9 (reference order)
#pragma unroll
    for (int nf = 0; nf < 4; ++nf)
#pragma unroll
      for (int r = 0; r < 8; ++r) {
        int qrow = qt * 64 + wave * 16 + hf * 8 + r;
        int key = kt * 64 + nf * 16 + nl;
        unsigned char mv = mask[((size_t)b * SEQ + qrow) * SEQ + key];
        float s = sc[nf][r];
        sc[nf][r] = mv ? s * 0.125f : -1e9f;
      }

    // online softmax; each row's 64 scores live across 16 lanes of one half
#pragma unroll
    for (int r = 0; r < 8; ++r) {
      float mx = fmaxf(fmaxf(sc[0][r], sc[1][r]), fmaxf(sc[2][r], sc[3][r]));
#pragma unroll
      for (int off = 8; off >= 1; off >>= 1) mx = fmaxf(mx, __shfl_xor(mx, off, 16));
      float mnew = fmaxf(m_i[r], mx);
      float alpha = __expf(m_i[r] - mnew);
      m_i[r] = mnew;
      float lsum = 0.f;
#pragma unroll
      for (int nf = 0; nf < 4; ++nf) {
        float p = __expf(sc[nf][r] - mnew);
        sc[nf][r] = p;
        lsum += p;
      }
#pragma unroll
      for (int off = 8; off >= 1; off >>= 1) lsum += __shfl_xor(lsum, off, 16);
      l_i[r] = l_i[r] * alpha + lsum;
#pragma unroll
      for (int nf = 0; nf < 4; ++nf) o_acc[nf][r] *= alpha;
    }

    // stage P (16x64) into LDS row-major for A-fragment reload
#pragma unroll
    for (int nf = 0; nf < 4; ++nf)
#pragma unroll
      for (int r = 0; r < 8; ++r)
        Ps[wave][(hf * 8 + r) * DHEAD + nf * 16 + nl] = (__bf16)sc[nf][r];
    __syncthreads();

    // O += P(16x64) . V(64x64)
    v16bf pa0 = load_a_frag(Ps[wave], 0, 0, DHEAD);
    v16bf pa1 = load_a_frag(Ps[wave], 0, 32, DHEAD);
#pragma unroll
    for (int nf = 0; nf < 4; ++nf) {
      v16bf vb0 = load_b_frag_km(Vs, 0, nf * 16, DHEAD);
      v16bf vb1 = load_b_frag_km(Vs, 32, nf * 16, DHEAD);
      o_acc[nf] = wmma_bf16(pa0, vb0, o_acc[nf]);
      o_acc[nf] = wmma_bf16(pa1, vb1, o_acc[nf]);
    }
  }

  // normalize and write pre-permuted (dd*H + h) so kernel 3 is a plain GEMM
#pragma unroll
  for (int nf = 0; nf < 4; ++nf)
#pragma unroll
    for (int r = 0; r < 8; ++r) {
      int qrow = qt * 64 + wave * 16 + hf * 8 + r;
      int dd = nf * 16 + nl;
      float o = o_acc[nf][r] / l_i[r];
      atn_ws[((size_t)b * SEQ + qrow) * DMODEL + dd * NHEADS + h] = (__bf16)o;
    }
}

// ---------------------------------------------------------------------------
// Kernel 3: out = atn @ Wout   (bf16 A, fp32 W -> fp32 out) M=4096 N=768 K=768
// ---------------------------------------------------------------------------
__global__ __launch_bounds__(128) void out_gemm_kernel(
    const __bf16* __restrict__ A, const float* __restrict__ W,
    float* __restrict__ out) {
  const int LDA = 40, LDB = 132;
  __shared__ __bf16 As[128 * 40];
  __shared__ __bf16 Bs[32 * 132];

  int tid = threadIdx.x, wave = tid >> 5, lane = tid & 31;
  int wm = wave >> 1, wn = wave & 1;
  int m_blk = blockIdx.y * 128, n_blk = blockIdx.x * 128;

  v8f zero = {0.f, 0.f, 0.f, 0.f, 0.f, 0.f, 0.f, 0.f};
  v8f acc[4][4];
#pragma unroll
  for (int i = 0; i < 4; ++i)
#pragma unroll
    for (int j = 0; j < 4; ++j) acc[i][j] = zero;

  for (int kt = 0; kt < DMODEL / 32; ++kt) {
    __syncthreads();
    if (kt + 1 < DMODEL / 32) {
      __builtin_prefetch(A + (size_t)(m_blk + tid) * DMODEL + (kt + 1) * 32, 0, 1);
      __builtin_prefetch(W + (size_t)((kt + 1) * 32 + (tid >> 2)) * DMODEL +
                             n_blk + ((tid & 3) << 5), 0, 1);
    }
#pragma unroll
    for (int it = 0; it < 32; ++it) {
      int idx = tid + it * 128;
      int r = idx >> 5, c = idx & 31;
      As[r * LDA + c] = A[(size_t)(m_blk + r) * DMODEL + kt * 32 + c];
    }
#pragma unroll
    for (int it = 0; it < 32; ++it) {
      int idx = tid + it * 128;
      int r = idx >> 7, c = idx & 127;
      Bs[r * LDB + c] = (__bf16)W[(size_t)(kt * 32 + r) * DMODEL + n_blk + c];
    }
    __syncthreads();

    v16bf af[4];
#pragma unroll
    for (int mf = 0; mf < 4; ++mf) af[mf] = load_a_frag(As, wm * 64 + mf * 16, 0, LDA);
#pragma unroll
    for (int nf = 0; nf < 4; ++nf) {
      v16bf bfr = load_b_frag_km(Bs, 0, wn * 64 + nf * 16, LDB);
#pragma unroll
      for (int mf = 0; mf < 4; ++mf) acc[mf][nf] = wmma_bf16(af[mf], bfr, acc[mf][nf]);
    }
  }

  int hf = lane >> 4, nl = lane & 15;
#pragma unroll
  for (int mf = 0; mf < 4; ++mf)
#pragma unroll
    for (int nf = 0; nf < 4; ++nf)
#pragma unroll
      for (int r = 0; r < 8; ++r) {
        int m = m_blk + wm * 64 + mf * 16 + hf * 8 + r;
        int n = n_blk + wn * 64 + nf * 16 + nl;
        out[(size_t)m * DMODEL + n] = acc[mf][nf][r];
      }
}

// ---------------------------------------------------------------------------
// Launch
// ---------------------------------------------------------------------------
extern "C" void kernel_launch(void* const* d_in, const int* in_sizes, int n_in,
                              void* d_out, int out_size, void* d_ws, size_t ws_size,
                              hipStream_t stream) {
  (void)in_sizes; (void)n_in; (void)out_size; (void)ws_size;
  const float* xs = (const float*)d_in[0];
  const unsigned char* mask = (const unsigned char*)d_in[1];
  const float* Wqkv = (const float*)d_in[2];
  const float* Wout = (const float*)d_in[3];
  float* out = (float*)d_out;

  const size_t per = (size_t)BATCH * NHEADS * SEQ * DHEAD;  // 3,145,728 elems
  char* ws = (char*)d_ws;
  __bf16* q_ws   = (__bf16*)(ws);
  __bf16* k_ws   = (__bf16*)(ws + per * 2);
  __bf16* v_ws   = (__bf16*)(ws + per * 4);
  __bf16* atn_ws = (__bf16*)(ws + per * 6);

  // 1) QKV projection: M=4096, N=2304 -> grid (18, 32)
  qkv_gemm_kernel<<<dim3(3 * DMODEL / 128, BATCH * SEQ / 128), 128, 0, stream>>>(
      xs, Wqkv, q_ws, k_ws, v_ws);

  // 2) Attention: B*H*(S/64) = 768 blocks of 128 threads
  attn_kernel<<<dim3(BATCH * NHEADS * (SEQ / 64)), 128, 0, stream>>>(
      q_ws, k_ws, v_ws, mask, atn_ws);

  // 3) Output projection: M=4096, N=768 -> grid (6, 32)
  out_gemm_kernel<<<dim3(DMODEL / 128, BATCH * SEQ / 128), 128, 0, stream>>>(
      atn_ws, Wout, out);
}
